// attention_22179211117298
// MI455X (gfx1250) — matrix operand order repaired and checked by reading, not compiled, not run on hardware
//
#include <hip/hip_runtime.h>

// ---------------------------------------------------------------------------
// MHA forward for MI455X (gfx1250, wave32).  All matmuls via
// v_wmma_f32_16x16x32_bf16 (bf16 inputs, f32 accumulate).
//
// Pipeline (all on `stream`):
//   1) q  = query_in  @ Wq          -> bf16 [N*L, E]           (ws +  0MB)
//   2) k  = keys_in   @ Wk^T        -> bf16 [N*L, E]           (ws + 16MB)
//   3) vT = (values_in@ Wv^T)^T     -> bf16 [N,H,D,L]          (ws + 32MB)
//   4) attn (flash, online softmax) -> bf16 [N*L, E]           (ws + 48MB)
//   5) out = attn @ Wo^T            -> f32  d_out
// Requires ws_size >= 64 MB.
// ---------------------------------------------------------------------------

typedef __attribute__((ext_vector_type(16))) __bf16 v16bf;
typedef __attribute__((ext_vector_type(8)))  float  v8f;

union FragU { unsigned int u[8]; v16bf v; };

__device__ __forceinline__ v8f wmma_acc(v16bf a, v16bf b, v8f c) {
  // D = A(16x32 bf16) * B(32x16 bf16) + C(16x16 f32)
  return __builtin_amdgcn_wmma_f32_16x16x32_bf16(
      /*neg_a=*/false, a, /*neg_b=*/false, b,
      /*c_mod=*/(short)0, c, /*reuse_a=*/false, /*reuse_b=*/false);
}

// A-fragment (16x32, 16-bit): lane l holds row m=l&15.
//   elems 0..15 -> K = (l>>4)*16 + 0..15
__device__ __forceinline__ v16bf load_afrag_g(const __bf16* base, int ld) {
  const int l = threadIdx.x & 31;
  const __bf16* row = base + (size_t)(l & 15) * ld + ((l >> 4) * 16);
  FragU f;
  const uint4 t0 = *(const uint4*)row;        // K klo..klo+7   (16B aligned)
  const uint4 t1 = *(const uint4*)(row + 8);  // K klo+8..+15
  f.u[0] = t0.x; f.u[1] = t0.y; f.u[2] = t0.z; f.u[3] = t0.w;
  f.u[4] = t1.x; f.u[5] = t1.y; f.u[6] = t1.z; f.u[7] = t1.w;
  return f.v;
}

__device__ __forceinline__ v16bf load_afrag_lds(const __bf16* base, int ld) {
  const int l = threadIdx.x & 31;
  const __bf16* row = base + (l & 15) * ld + ((l >> 4) * 16);
  FragU f;
#pragma unroll
  for (int p = 0; p < 4; ++p) {            // 4B-aligned pair reads
    f.u[p]     = *(const unsigned int*)(row + 2 * p);
    f.u[4 + p] = *(const unsigned int*)(row + 8 + 2 * p);
  }
  return f.v;
}

// B-fragment (32x16): lane l holds column n=l&15, K = (l>>4)*16 + 0..15,
// read from a source laid out n-major with contiguous K (i.e. B^T rows).
__device__ __forceinline__ v16bf load_bfrag_g(const __bf16* base, int ld) {
  const int l = threadIdx.x & 31;
  const __bf16* p = base + (size_t)(l & 15) * ld + ((l >> 4) * 16);
  FragU f;
  const uint4 t0 = *(const uint4*)p;
  const uint4 t1 = *(const uint4*)(p + 8);
  f.u[0] = t0.x; f.u[1] = t0.y; f.u[2] = t0.z; f.u[3] = t0.w;
  f.u[4] = t1.x; f.u[5] = t1.y; f.u[6] = t1.z; f.u[7] = t1.w;
  return f.v;
}

__device__ __forceinline__ v16bf load_bfrag_ldsT(const __bf16* base, int ld) {
  const int l = threadIdx.x & 31;
  const __bf16* p = base + (l & 15) * ld + ((l >> 4) * 16);
  FragU f;
#pragma unroll
  for (int i = 0; i < 8; ++i) f.u[i] = *(const unsigned int*)(p + 2 * i);
  return f.v;
}

__device__ __forceinline__ float redmax16(float v) {
  v = fmaxf(v, __shfl_xor(v, 1));
  v = fmaxf(v, __shfl_xor(v, 2));
  v = fmaxf(v, __shfl_xor(v, 4));
  v = fmaxf(v, __shfl_xor(v, 8));
  return v;
}
__device__ __forceinline__ float redsum16(float v) {
  v += __shfl_xor(v, 1); v += __shfl_xor(v, 2);
  v += __shfl_xor(v, 4); v += __shfl_xor(v, 8);
  return v;
}

// ---------------------------------------------------------------------------
// Generic GEMM:  OUT[M=8192, N=1024] = X[M,1024] * B  with
//   B(k,n) = TRANSW ? W[n*1024+k] : W[k*1024+n]
// Block tile 128x64, 8 waves, wave tile 32x32 (2x2 WMMA), K-step 64
// (two 32-deep WMMA sub-steps per barrier pair -> 8 WMMA / stage).
// ---------------------------------------------------------------------------
template <bool XBF16, bool TRANSW, bool OUTF32, bool STOREVT>
__global__ __launch_bounds__(256) void gemm_bf16_wmma_kernel(
    const void* __restrict__ Xv, const float* __restrict__ W,
    void* __restrict__ OUTv) {
  constexpr int E = 1024;
  constexpr int BK = 64;
  constexpr int LDA = BK + 4;        // 68 bf16 = 136B row stride
  __shared__ __bf16 sA[128 * LDA];   // A tile 128x64 (+pad)   ~17.4KB
  __shared__ __bf16 sBt[64 * LDA];   // B tile transposed [n][k] ~8.7KB
  const int tid = threadIdx.x;
  const int rowBase = blockIdx.y * 128;
  const int colBase = blockIdx.x * 64;
  const int wave = tid >> 5;
  const int rowOff = (wave >> 1) * 32;
  const int colOff = (wave & 1) * 32;
  v8f acc[2][2] = {};

  for (int k0 = 0; k0 < E; k0 += BK) {
    // ---- stage A tile (convert f32 -> bf16 when needed) ----
    if (XBF16) {
      const __bf16* X = (const __bf16*)Xv;
#pragma unroll
      for (int i = 0; i < 4; ++i) {
        const int flat = (tid + i * 256) * 8;          // 8192 bf16 total
        const int row = flat >> 6, col = flat & 63;
        const __bf16* src = X + (size_t)(rowBase + row) * E + k0 + col;
        const uint4 t = *(const uint4*)src;
        if (k0 + BK < E) __builtin_prefetch(src + BK, 0, 3);
        unsigned int* d = (unsigned int*)(sA + row * LDA + col);
        d[0] = t.x; d[1] = t.y; d[2] = t.z; d[3] = t.w;
      }
    } else {
      const float* X = (const float*)Xv;
#pragma unroll
      for (int i = 0; i < 8; ++i) {
        const int flat = (tid + i * 256) * 4;          // 8192 f32 total
        const int row = flat >> 6, col = flat & 63;
        const float* src = X + (size_t)(rowBase + row) * E + k0 + col;
        const float4 t = *(const float4*)src;
        if (k0 + BK < E) __builtin_prefetch(src + BK, 0, 3);
        __bf16* d = sA + row * LDA + col;
        d[0] = (__bf16)t.x; d[1] = (__bf16)t.y;
        d[2] = (__bf16)t.z; d[3] = (__bf16)t.w;
      }
    }
    // ---- stage B tile transposed: sBt[n][k] = B(k0+k, colBase+n) ----
#pragma unroll
    for (int i = 0; i < 4; ++i) {
      const int flat = (tid + i * 256) * 4;            // 4096 f32 total
      if (TRANSW) {
        const int n = flat >> 6, kk = flat & 63;
        const float* src = W + (size_t)(colBase + n) * E + k0 + kk;
        const float4 t = *(const float4*)src;
        if (k0 + BK < E) __builtin_prefetch(src + BK, 0, 3);
        __bf16* d = sBt + n * LDA + kk;
        d[0] = (__bf16)t.x; d[1] = (__bf16)t.y;
        d[2] = (__bf16)t.z; d[3] = (__bf16)t.w;
      } else {
        const int kk = flat >> 6, n = flat & 63;
        const float* src = W + (size_t)(k0 + kk) * E + colBase + n;
        const float4 t = *(const float4*)src;
        if (k0 + BK < E) __builtin_prefetch(src + (size_t)BK * E, 0, 3);
        sBt[(n + 0) * LDA + kk] = (__bf16)t.x;
        sBt[(n + 1) * LDA + kk] = (__bf16)t.y;
        sBt[(n + 2) * LDA + kk] = (__bf16)t.z;
        sBt[(n + 3) * LDA + kk] = (__bf16)t.w;
      }
    }
    __syncthreads();

#pragma unroll
    for (int ks = 0; ks < 2; ++ks) {                   // two 32-deep sub-steps
      const v16bf a0 = load_afrag_lds(sA + rowOff * LDA + ks * 32, LDA);
      const v16bf a1 = load_afrag_lds(sA + (rowOff + 16) * LDA + ks * 32, LDA);
      const v16bf b0 = load_bfrag_ldsT(sBt + colOff * LDA + ks * 32, LDA);
      const v16bf b1 = load_bfrag_ldsT(sBt + (colOff + 16) * LDA + ks * 32, LDA);
      acc[0][0] = wmma_acc(a0, b0, acc[0][0]);
      acc[0][1] = wmma_acc(a0, b1, acc[0][1]);
      acc[1][0] = wmma_acc(a1, b0, acc[1][0]);
      acc[1][1] = wmma_acc(a1, b1, acc[1][1]);
    }
    __syncthreads();
  }

  // ---- epilogue: C layout (m = vgpr + 8*(lane>>4), n = lane&15) ----
  const int l = tid & 31, nn = l & 15, half = l >> 4;
#pragma unroll
  for (int ti = 0; ti < 2; ++ti) {
#pragma unroll
    for (int tj = 0; tj < 2; ++tj) {
      const int gm0 = rowBase + rowOff + ti * 16 + half * 8;
      const int gn = colBase + colOff + tj * 16 + nn;
#pragma unroll
      for (int r = 0; r < 8; ++r) {
        const float val = acc[ti][tj][r];
        const int gm = gm0 + r;
        if (OUTF32) {
          ((float*)OUTv)[(size_t)gm * E + gn] = val;
        } else if (STOREVT) {
          // (gm -> batch,key ; gn -> head,d)  => vT[batch][head][d][key]
          const int batch = gm >> 10, key = gm & 1023;
          const int hh = gn >> 6, dd = gn & 63;
          ((__bf16*)OUTv)[((size_t)((batch * 16 + hh) * 64 + dd) << 10) + key] =
              (__bf16)val;
        } else {
          ((__bf16*)OUTv)[(size_t)gm * E + gn] = (__bf16)val;
        }
      }
    }
  }
}

// ---------------------------------------------------------------------------
// Flash attention: one wave = (batch, head, 32-row q tile), KV step = 32.
// 32 q-rows per wave reuse every K/V B-fragment twice -> halves K/V traffic
// (16 WMMA per step against 8 B-fragment loads).
// ---------------------------------------------------------------------------
__global__ __launch_bounds__(256) void attn_fwd_kernel(
    const __bf16* __restrict__ Q, const __bf16* __restrict__ K,
    const __bf16* __restrict__ VT, const unsigned char* __restrict__ mq,
    const unsigned char* __restrict__ mk, __bf16* __restrict__ Obuf) {
  constexpr int E = 1024, L = 1024, D = 64;
  constexpr int LDP = 34;                       // P buffer row stride (bf16)
  __shared__ __bf16 sP[8 * 2 * 16 * LDP];       // per-wave 2x(16x32) P tiles
  const int tid = threadIdx.x;
  const int wave = tid >> 5;
  const int l = tid & 31, nn = l & 15, half = l >> 4;
  __bf16* Pw = sP + wave * (2 * 16 * LDP);

  const int w = blockIdx.x * 8 + wave;          // 4096 waves total
  const int qt = w & 31;                        // 32-row q tile
  const int h = (w >> 5) & 15;
  const int b = w >> 9;

  const __bf16* Qb = Q + ((size_t)(b * L + qt * 32) * E + h * D);
  const __bf16* Kb = K + ((size_t)(b * L) * E + h * D);
  const __bf16* Vb = VT + (size_t)((b * 16 + h) * D) * L;

  v16bf qa[2][2];
#pragma unroll
  for (int i = 0; i < 2; ++i) {
    qa[i][0] = load_afrag_g(Qb + (size_t)(i * 16) * E, E);       // d 0..31
    qa[i][1] = load_afrag_g(Qb + (size_t)(i * 16) * E + 32, E);  // d 32..63
  }

  float mi[2][8], li[2][8];
  bool qm[2][8];
  v8f O[2][4] = {};
  const unsigned char* mqRow = mq + b * L + qt * 32;
  const unsigned char* mkRow = mk + b * L;
#pragma unroll
  for (int i = 0; i < 2; ++i)
#pragma unroll
    for (int r = 0; r < 8; ++r) {
      mi[i][r] = -1e30f; li[i][r] = 0.f;
      qm[i][r] = (mqRow[i * 16 + r + half * 8] != 0);
    }

  const float scale = 0.03125f;                 // 1/sqrt(E) = 1/32

  for (int kt = 0; kt < L; kt += 32) {
    if (kt + 32 < L) {                          // global_prefetch_b8 next tile
      __builtin_prefetch(Kb + (size_t)(kt + 32 + nn) * E + half * 32, 0, 3);
      __builtin_prefetch(Vb + (size_t)(half * 32 + nn) * L + kt + 32, 0, 3);
    }
    // K B-fragments, shared by both q row-tiles.
    const v16bf kb0a = load_bfrag_g(Kb + (size_t)kt * E, E);
    const v16bf kb0b = load_bfrag_g(Kb + (size_t)kt * E + 32, E);
    const v16bf kb1a = load_bfrag_g(Kb + (size_t)(kt + 16) * E, E);
    const v16bf kb1b = load_bfrag_g(Kb + (size_t)(kt + 16) * E + 32, E);
    const bool km0 = (mkRow[kt + nn] != 0);
    const bool km1 = (mkRow[kt + 16 + nn] != 0);

#pragma unroll
    for (int i = 0; i < 2; ++i) {
      v8f s0 = {}, s1 = {};
      s0 = wmma_acc(qa[i][0], kb0a, s0);
      s0 = wmma_acc(qa[i][1], kb0b, s0);
      s1 = wmma_acc(qa[i][0], kb1a, s1);
      s1 = wmma_acc(qa[i][1], kb1b, s1);
#pragma unroll
      for (int r = 0; r < 8; ++r) {
        const float sa = (qm[i][r] && km0) ? s0[r] * scale : -1e30f;
        const float sb = (qm[i][r] && km1) ? s1[r] * scale : -1e30f;
        const float rmax = redmax16(fmaxf(sa, sb));
        const float mnew = fmaxf(mi[i][r], rmax);
        const float alpha = __expf(mi[i][r] - mnew);
        const float p0 = __expf(sa - mnew);
        const float p1 = __expf(sb - mnew);
        const float rsum = redsum16(p0 + p1);
        li[i][r] = li[i][r] * alpha + rsum;
        mi[i][r] = mnew;
        O[i][0][r] *= alpha; O[i][1][r] *= alpha;
        O[i][2][r] *= alpha; O[i][3][r] *= alpha;
        const int m = r + half * 8;
        Pw[(i * 16 + m) * LDP + nn] = (__bf16)p0;        // local keys 0..15
        Pw[(i * 16 + m) * LDP + 16 + nn] = (__bf16)p1;   // local keys 16..31
      }
    }
    // wave-private LDS relayout C->A: wait only our DS stores (no barrier).
    asm volatile("s_wait_dscnt 0" ::: "memory");
    v16bf vb[4];
#pragma unroll
    for (int j = 0; j < 4; ++j)
      vb[j] = load_bfrag_g(Vb + (size_t)(j * 16) * L + kt, L);
#pragma unroll
    for (int i = 0; i < 2; ++i) {
      const v16bf pa = load_afrag_lds(Pw + i * 16 * LDP, LDP);
#pragma unroll
      for (int j = 0; j < 4; ++j) O[i][j] = wmma_acc(pa, vb[j], O[i][j]);
    }
  }

  __bf16* Ob = Obuf + ((size_t)(b * L + qt * 32) * E + h * D);
#pragma unroll
  for (int i = 0; i < 2; ++i)
#pragma unroll
    for (int j = 0; j < 4; ++j)
#pragma unroll
      for (int r = 0; r < 8; ++r) {
        float val = O[i][j][r] / li[i][r];
        if (!qm[i][r]) val = 0.f;
        Ob[(size_t)(i * 16 + r + half * 8) * E + j * 16 + nn] = (__bf16)val;
      }
}

// ---------------------------------------------------------------------------
extern "C" void kernel_launch(void* const* d_in, const int* in_sizes, int n_in,
                              void* d_out, int out_size, void* d_ws,
                              size_t ws_size, hipStream_t stream) {
  (void)in_sizes; (void)n_in; (void)out_size; (void)ws_size;
  const float* values_in = (const float*)d_in[0];
  const float* keys_in   = (const float*)d_in[1];
  const float* query_in  = (const float*)d_in[2];
  const unsigned char* mask_k = (const unsigned char*)d_in[3];  // bool, 1B
  const unsigned char* mask_q = (const unsigned char*)d_in[4];
  const float* Wv = (const float*)d_in[5];
  const float* Wk = (const float*)d_in[6];
  const float* Wq = (const float*)d_in[7];
  const float* Wo = (const float*)d_in[8];

  const size_t NLE = (size_t)8192 * 1024;       // 8.4M bf16 = 16MB each
  __bf16* qws  = (__bf16*)d_ws;
  __bf16* kws  = qws + NLE;
  __bf16* vtws = kws + NLE;
  __bf16* aws  = vtws + NLE;

  const dim3 gGemm(16, 64);                      // N/64 x M/128
  const dim3 blk(256);

  // q = query_in @ Wq             (no transpose)
  gemm_bf16_wmma_kernel<false, false, false, false>
      <<<gGemm, blk, 0, stream>>>(query_in, Wq, qws);
  // k = keys_in @ Wk^T
  gemm_bf16_wmma_kernel<false, true, false, false>
      <<<gGemm, blk, 0, stream>>>(keys_in, Wk, kws);
  // vT = (values_in @ Wv^T)^T  stored [N,H,D,L]
  gemm_bf16_wmma_kernel<false, true, false, true>
      <<<gGemm, blk, 0, stream>>>(values_in, Wv, vtws);
  // attention: 4096 waves = 8 batches x 16 heads x 32 q-tiles (32 rows)
  attn_fwd_kernel<<<dim3(512), blk, 0, stream>>>(qws, kws, vtws, mask_q,
                                                 mask_k, aws);
  // out = attn @ Wo^T  (f32 output)
  gemm_bf16_wmma_kernel<true, true, true, false>
      <<<gGemm, blk, 0, stream>>>(aws, Wo, (float*)d_out);
}
